// EncoderVQVAEFixed_72756745994826
// MI455X (gfx1250) — matrix operand order repaired and checked by reading, not compiled, not run on hardware
//
#include <hip/hip_runtime.h>
#include <hip/hip_bf16.h>

typedef __bf16 bf16;
typedef __attribute__((ext_vector_type(16))) __bf16 v16bf;
typedef __attribute__((ext_vector_type(8)))  __bf16 v8bf;
typedef __attribute__((ext_vector_type(8)))  float  v8f;

#define TILE_M 64
#define TILE_N 128
#define TILE_K 32
#define LDA_S  40    // A tile row pad: 40 bf16 = 80 B (16B aligned, 20-dword bank stride)
#define LDB_S  40    // transposed B tile [col][k] row pad: 80 B

#define B_SZ     256
#define ENC_DIM  768
#define LATENT   128
#define KCODES   512
#define IN_DIM   27000
#define RECON_N  (B_SZ * IN_DIM)            // 6,912,000
#define N_TILES5 ((IN_DIM + TILE_N - 1) / TILE_N)   // 211
#define SSE_PARTS (N_TILES5 * (B_SZ / TILE_M))      // 844 partials from gemm5

// ---------------------------------------------------------------------------
// Fragment loads per CDNA5 WMMA VGPR layout (two aligned 16B LDS reads each).
// A 16x32 bf16: lane<16 holds K=0..7 & 16..23; lane>=16 holds K=8..15 & 24..31.
// B 32x16 bf16: lane<16 holds K=0..7 & 16..23; lane>=16 holds K=8..15 & 24..31 (col = lane&15).
// B tile is stored transposed in LDS: sBt[col][k] -> contiguous K per lane.
// ---------------------------------------------------------------------------
__device__ __forceinline__ v16bf load_afrag(const bf16 (*sA)[LDA_S], int row, int half) {
    const v8bf lo = *(const v8bf*)&sA[row][half * 8];
    const v8bf hi = *(const v8bf*)&sA[row][16 + half * 8];
    return __builtin_shufflevector(lo, hi, 0,1,2,3,4,5,6,7,8,9,10,11,12,13,14,15);
}
__device__ __forceinline__ v16bf load_bfrag(const bf16 (*sBt)[LDB_S], int col, int half) {
    const v8bf lo = *(const v8bf*)&sBt[col][half * 8];
    const v8bf hi = *(const v8bf*)&sBt[col][16 + half * 8];
    return __builtin_shufflevector(lo, hi, 0,1,2,3,4,5,6,7,8,9,10,11,12,13,14,15);
}

struct StageRegs { float4 a0, a1, b0, b1, b2, b3; };

__device__ __forceinline__ void fast_load(StageRegs& r, const float* __restrict__ A,
                                          const float* __restrict__ B,
                                          int rowBase, int colBase, int k0,
                                          int K, int N, int tid) {
    const int ar = tid >> 2, ac = (tid & 3) * 8;
    const float* ap = A + (long)(rowBase + ar) * K + k0 + ac;
    r.a0 = *(const float4*)ap;
    r.a1 = *(const float4*)(ap + 4);
    const int br = tid >> 3, bc = (tid & 7) * 16;
    const float* bp = B + (long)(k0 + br) * N + colBase + bc;
    r.b0 = *(const float4*)(bp);
    r.b1 = *(const float4*)(bp + 4);
    r.b2 = *(const float4*)(bp + 8);
    r.b3 = *(const float4*)(bp + 12);
}

__device__ __forceinline__ void fast_store(const StageRegs& r,
                                           bf16 (*sA)[LDA_S], bf16 (*sBt)[LDB_S], int tid) {
    const int ar = tid >> 2, ac = (tid & 3) * 8;
    v8bf av;
    av[0] = (bf16)r.a0.x; av[1] = (bf16)r.a0.y; av[2] = (bf16)r.a0.z; av[3] = (bf16)r.a0.w;
    av[4] = (bf16)r.a1.x; av[5] = (bf16)r.a1.y; av[6] = (bf16)r.a1.z; av[7] = (bf16)r.a1.w;
    *(v8bf*)&sA[ar][ac] = av;
    const int br = tid >> 3, bc = (tid & 7) * 16;
    const float bv[16] = { r.b0.x, r.b0.y, r.b0.z, r.b0.w,  r.b1.x, r.b1.y, r.b1.z, r.b1.w,
                           r.b2.x, r.b2.y, r.b2.z, r.b2.w,  r.b3.x, r.b3.y, r.b3.z, r.b3.w };
    #pragma unroll
    for (int j = 0; j < 16; ++j) sBt[bc + j][br] = (bf16)bv[j];
}

// cold path: element-guarded staging (K remainder step / N edge blocks only)
__device__ __forceinline__ void stage_guarded(const float* __restrict__ A,
                                              const float* __restrict__ B,
                                              bf16 (*sA)[LDA_S], bf16 (*sBt)[LDB_S],
                                              int rowBase, int colBase, int k0,
                                              int K, int N, int tid) {
    const int ar = tid >> 2, ac = (tid & 3) * 8;
    #pragma unroll
    for (int j = 0; j < 8; ++j) {
        int gc = k0 + ac + j;
        float v = (gc < K) ? A[(long)(rowBase + ar) * K + gc] : 0.0f;
        sA[ar][ac + j] = (bf16)v;
    }
    const int br = tid >> 3, bc = (tid & 7) * 16;
    const int gr = k0 + br;
    #pragma unroll
    for (int j = 0; j < 16; ++j) {
        int gc = colBase + bc + j;
        float v = (gr < K && gc < N) ? B[(long)gr * N + gc] : 0.0f;
        sBt[bc + j][br] = (bf16)v;
    }
}

// ---------------------------------------------------------------------------
// C[M,N] = A[M,K] @ B[K,N] (+bias, +relu), fp32 I/O, bf16 WMMA, fp32 accum.
// 256 threads = 8 waves (2Mx4N); wave tile 32x32 = 2x2 v_wmma_f32_16x16x32_bf16.
// Double-buffered LDS pipeline; M must be a multiple of 64.
// Optional fused SSE: if sse_out != nullptr, also reduces sum((C - Xref)^2)
// over this block's tile into sse_out[blockIdx.y * gridDim.x + blockIdx.x].
// ---------------------------------------------------------------------------
__global__ __launch_bounds__(256)
void gemm_bf16_wmma(const float* __restrict__ A, const float* __restrict__ B,
                    const float* __restrict__ bias, float* __restrict__ C,
                    int M, int N, int K, int relu,
                    const float* __restrict__ Xref, float* __restrict__ sse_out)
{
    __shared__ alignas(16) bf16 sA[2][TILE_M][LDA_S];   // 10 KB
    __shared__ alignas(16) bf16 sBt[2][TILE_N][LDB_S];  // 20 KB, transposed [col][k]
    __shared__ float sred[256];                          // fused-SSE reduction

    const int tid  = threadIdx.x;
    const int lane = tid & 31;
    const int wave = tid >> 5;
    const int wm   = (wave & 1) * 32;
    const int wn   = (wave >> 1) * 32;

    const int rowBase = blockIdx.y * TILE_M;
    const int colBase = blockIdx.x * TILE_N;
    const bool edgeN  = (colBase + TILE_N > N);

    const int row16 = lane & 15;
    const int half  = lane >> 4;

    v8f c00 = {}, c01 = {}, c10 = {}, c11 = {};

    auto compute = [&](int buf) {
        v16bf a0 = load_afrag(sA[buf], wm + row16, half);
        v16bf a1 = load_afrag(sA[buf], wm + 16 + row16, half);
        v16bf b0 = load_bfrag(sBt[buf], wn + row16, half);
        v16bf b1 = load_bfrag(sBt[buf], wn + 16 + row16, half);
        c00 = __builtin_amdgcn_wmma_f32_16x16x32_bf16(false, a0, false, b0, (short)0, c00, false, false);
        c01 = __builtin_amdgcn_wmma_f32_16x16x32_bf16(false, a0, false, b1, (short)0, c01, false, false);
        c10 = __builtin_amdgcn_wmma_f32_16x16x32_bf16(false, a1, false, b0, (short)0, c10, false, false);
        c11 = __builtin_amdgcn_wmma_f32_16x16x32_bf16(false, a1, false, b1, (short)0, c11, false, false);
    };

    const int kfull = K / TILE_K;
    const int krem  = K - kfull * TILE_K;

    if (!edgeN) {
        // ---- fast path: unguarded vectorized staging, double-buffered ----
        StageRegs regs;
        fast_load(regs, A, B, rowBase, colBase, 0, K, N, tid);
        fast_store(regs, sA[0], sBt[0], tid);
        int cur = 0;
        for (int ks = 0; ks < kfull; ++ks) {
            if (ks + 1 < kfull)
                fast_load(regs, A, B, rowBase, colBase, (ks + 1) * TILE_K, K, N, tid);
            if (ks + 2 < kfull)
                __builtin_prefetch(B + (long)((ks + 2) * TILE_K + (tid >> 3)) * N
                                     + colBase + (tid & 7) * 16, 0, 0);
            __syncthreads();                 // buf[cur] visible to all waves
            compute(cur);
            if (ks + 1 < kfull)
                fast_store(regs, sA[cur ^ 1], sBt[cur ^ 1], tid);
            cur ^= 1;
        }
        if (krem) {
            __syncthreads();                 // prior reads of buf0 done
            stage_guarded(A, B, sA[0], sBt[0], rowBase, colBase, kfull * TILE_K, K, N, tid);
            __syncthreads();
            compute(0);
        }
    } else {
        // ---- edge path: element-guarded staging every step ----
        const int ksteps = (K + TILE_K - 1) / TILE_K;
        for (int ks = 0; ks < ksteps; ++ks) {
            __syncthreads();
            stage_guarded(A, B, sA[0], sBt[0], rowBase, colBase, ks * TILE_K, K, N, tid);
            __syncthreads();
            compute(0);
        }
    }

    // ---- writeback: C/D layout row = r + 8*half, col = lane&15 per 16x16 tile ----
    float ssep = 0.0f;
    auto store_tile = [&](v8f& acc, int mt, int nt) {
        const int col = colBase + wn + nt + row16;
        if (!edgeN || col < N) {
            float bvv = bias ? bias[col] : 0.0f;
            #pragma unroll
            for (int r = 0; r < 8; ++r) {
                const int row = rowBase + wm + mt + half * 8 + r;  // M multiple of 64
                float v = acc[r] + bvv;
                if (relu) v = fmaxf(v, 0.0f);
                C[(long)row * N + col] = v;
                if (Xref) {
                    float d = v - Xref[(long)row * N + col];
                    ssep = fmaf(d, d, ssep);
                }
            }
        }
    };
    store_tile(c00, 0, 0);
    store_tile(c01, 0, 16);
    store_tile(c10, 16, 0);
    store_tile(c11, 16, 16);

    if (sse_out) {
        __syncthreads();                    // LDS buffers free for reuse
        sred[tid] = ssep;
        __syncthreads();
        for (int s = 128; s > 0; s >>= 1) {
            if (tid < s) sred[tid] += sred[tid + s];
            __syncthreads();
        }
        if (tid == 0) sse_out[blockIdx.y * gridDim.x + blockIdx.x] = sred[0];
    }
}

// ---------------------------------------------------------------------------
// VQ: one block per batch row; squared distance to 512 codes, argmin with
// first-index tiebreak (matches jnp.argmin), gather, per-row (q-z)^2 partial.
// ---------------------------------------------------------------------------
__global__ __launch_bounds__(256)
void vq_kernel(const float* __restrict__ z, const float* __restrict__ cb,
               float* __restrict__ z_q, int* __restrict__ idx_out,
               float* __restrict__ vq_part)
{
    __shared__ float zrow[LATENT];
    __shared__ float sd[256];
    __shared__ int   si[256];

    const int b = blockIdx.x;
    const int tid = threadIdx.x;

    if (tid < LATENT) zrow[tid] = z[(long)b * LATENT + tid];
    __syncthreads();

    float best = 3.4e38f;
    int   bk   = 0x7fffffff;
    for (int k = tid; k < KCODES; k += 256) {
        const float* e = cb + (long)k * LATENT;
        float d = 0.0f;
        #pragma unroll 4
        for (int j = 0; j < LATENT; ++j) {
            float t = zrow[j] - e[j];
            d = fmaf(t, t, d);
        }
        if (d < best || (d == best && k < bk)) { best = d; bk = k; }
    }
    sd[tid] = best; si[tid] = bk;
    __syncthreads();
    for (int s = 128; s > 0; s >>= 1) {
        if (tid < s) {
            float d2 = sd[tid + s]; int k2 = si[tid + s];
            if (d2 < sd[tid] || (d2 == sd[tid] && k2 < si[tid])) { sd[tid] = d2; si[tid] = k2; }
        }
        __syncthreads();
    }
    const int kb = si[0];
    __syncthreads();

    float p = 0.0f;
    if (tid < LATENT) {
        float q = cb[(long)kb * LATENT + tid];
        z_q[(long)b * LATENT + tid] = q;      // straight-through forward value == quant
        float t = q - zrow[tid];
        p = t * t;
    }
    sd[tid] = p;
    __syncthreads();
    for (int s = 128; s > 0; s >>= 1) {
        if (tid < s) sd[tid] += sd[tid + s];
        __syncthreads();
    }
    if (tid == 0) { idx_out[b] = kb; vq_part[b] = sd[0]; }
}

// out: [0..RECON_N) x_recon | [RECON_N] total loss | [RECON_N+1] vq_loss | +256 idx
__global__ __launch_bounds__(256)
void finalize_kernel(const float* __restrict__ sse_part, int nparts,
                     const float* __restrict__ vq_part,
                     const int* __restrict__ idx,
                     float* __restrict__ out)
{
    __shared__ float s1[256];
    __shared__ float s2[256];
    const int tid = threadIdx.x;

    float a = 0.0f;
    for (int i = tid; i < nparts; i += 256) a += sse_part[i];
    s1[tid] = a;
    s2[tid] = vq_part[tid];
    __syncthreads();
    for (int s = 128; s > 0; s >>= 1) {
        if (tid < s) { s1[tid] += s1[tid + s]; s2[tid] += s2[tid + s]; }
        __syncthreads();
    }
    if (tid == 0) {
        float recon_loss = s1[0] / (float)RECON_N;
        float vq_loss = (1.0f + 0.1f) * (s2[0] / (float)(B_SZ * LATENT));
        out[RECON_N]     = recon_loss + vq_loss;
        out[RECON_N + 1] = vq_loss;
    }
    out[RECON_N + 2 + tid] = (float)idx[tid];
}

// ---------------------------------------------------------------------------
extern "C" void kernel_launch(void* const* d_in, const int* in_sizes, int n_in,
                              void* d_out, int out_size, void* d_ws, size_t ws_size,
                              hipStream_t stream)
{
    const float* x     = (const float*)d_in[0];
    const float* W_enc = (const float*)d_in[1];
    const float* W_lat = (const float*)d_in[2];
    const float* b_lat = (const float*)d_in[3];
    const float* cb    = (const float*)d_in[4];
    const float* W_d1  = (const float*)d_in[5];
    const float* b_d1  = (const float*)d_in[6];
    const float* W_d2  = (const float*)d_in[7];
    const float* b_d2  = (const float*)d_in[8];
    float* out = (float*)d_out;

    // workspace layout (floats)
    float* ws        = (float*)d_ws;
    float* features  = ws;                       // 256*768 = 196608 (reused for h)
    float* z         = ws + 196608;              // 256*128 =  32768
    float* z_q       = ws + 229376;              // 256*128 =  32768
    int*   codes     = (int*)(ws + 262144);      // 256
    float* vq_part   = ws + 262400;              // 256
    float* sse_part  = ws + 262656;              // 844

    const dim3 blk(256);

    // 1) features = X @ W_enc            [256,27000]x[27000,768]
    gemm_bf16_wmma<<<dim3(ENC_DIM / TILE_N, B_SZ / TILE_M), blk, 0, stream>>>(
        x, W_enc, nullptr, features, B_SZ, ENC_DIM, IN_DIM, 0, nullptr, nullptr);

    // 2) z = features @ W_lat + b_lat    [256,768]x[768,128]
    gemm_bf16_wmma<<<dim3(1, B_SZ / TILE_M), blk, 0, stream>>>(
        features, W_lat, b_lat, z, B_SZ, LATENT, ENC_DIM, 0, nullptr, nullptr);

    // 3) vector quantization
    vq_kernel<<<B_SZ, blk, 0, stream>>>(z, cb, z_q, codes, vq_part);

    // 4) h = relu(z_q @ W_d1 + b_d1)     [256,128]x[128,768]  (reuse features buf)
    gemm_bf16_wmma<<<dim3(ENC_DIM / TILE_N, B_SZ / TILE_M), blk, 0, stream>>>(
        z_q, W_d1, b_d1, features, B_SZ, ENC_DIM, LATENT, 1, nullptr, nullptr);

    // 5) x_recon = h @ W_d2 + b_d2 -> out[0..RECON_N), fused (x_recon - x)^2 partials
    gemm_bf16_wmma<<<dim3(N_TILES5, B_SZ / TILE_M), blk, 0, stream>>>(
        features, W_d2, b_d2, out, B_SZ, IN_DIM, ENC_DIM, 0, x, sse_part);

    // 6) losses + indices
    finalize_kernel<<<1, blk, 0, stream>>>(sse_part, SSE_PARTS, vq_part, codes, out);
}
